// MyGNN_66254165508911
// MI455X (gfx1250) — matrix operand order repaired and checked by reading, not compiled, not run on hardware
//
#include <hip/hip_runtime.h>
#include <hip/hip_bf16.h>

typedef _Float16 half16 __attribute__((ext_vector_type(16)));
typedef _Float16 half8  __attribute__((ext_vector_type(8)));
typedef float    floatx8 __attribute__((ext_vector_type(8)));

#define Bc 256
#define Nc 64
#define Mrows (Bc * Nc)      // 16384
#define LDX 104              // f32 activation stride (max fin=100)
#define LDH 224              // f16 augmented-feature stride (max in_aug=201 -> 224)
#define EPSBN 1e-3f

// ---------------------------------------------------------------------------
// Kernel 0: embeddings + concat -> x0 (f32, stride LDX), and mask
// ---------------------------------------------------------------------------
__global__ void k_embed(const float* __restrict__ xx, const int* __restrict__ xcat,
                        const float* __restrict__ e0, const float* __restrict__ e1,
                        const float* __restrict__ e2, const float* __restrict__ e3,
                        float* __restrict__ xbuf, float* __restrict__ mask) {
  int idx = blockIdx.x * blockDim.x + threadIdx.x;
  if (idx >= Mrows) return;
  const float* src = xx + (size_t)idx * 36;
  float* dst = xbuf + (size_t)idx * LDX;
  #pragma unroll
  for (int f = 0; f < 36; ++f) dst[f] = src[f];
  const int* c = xcat + (size_t)idx * 4;
  dst[36] = e0[c[0]*2];  dst[37] = e0[c[0]*2+1];
  dst[38] = e1[c[1]*2];  dst[39] = e1[c[1]*2+1];
  dst[40] = e2[c[2]*2];  dst[41] = e2[c[2]*2+1];
  dst[42] = e3[c[3]*2];  dst[43] = e3[c[3]*2+1];
  mask[idx] = src[0];
}

// ---------------------------------------------------------------------------
// Kernel 1: pairwise RBF sums. One block per batch, 64 threads (2 waves).
// wsum[row] = mask * sum_j exp(-10*||c_j-c_n||^2) ; sdist[row] = mask * sum_j w*d
// ---------------------------------------------------------------------------
__global__ void k_pairwise(const float* __restrict__ xbuf, const float* __restrict__ mask,
                           int fin, int ndim,
                           float* __restrict__ wsum, float* __restrict__ sdist) {
  __shared__ float sc[Nc * 4];
  int b = blockIdx.x, n = threadIdx.x;
  int row = b * Nc + n;
  const float* xr = xbuf + (size_t)row * LDX + (fin - ndim);
  #pragma unroll
  for (int d = 0; d < 4; ++d) sc[n * 4 + d] = (d < ndim) ? xr[d] : 0.0f;
  __syncthreads();
  float c0 = sc[n*4+0], c1 = sc[n*4+1], c2 = sc[n*4+2], c3 = sc[n*4+3];
  float ws = 0.0f, sd = 0.0f;
  for (int j = 0; j < Nc; ++j) {
    float d0 = sc[j*4+0] - c0, d1 = sc[j*4+1] - c1;
    float d2 = sc[j*4+2] - c2, d3 = sc[j*4+3] - c3;
    float dist = d0*d0 + d1*d1 + d2*d2 + d3*d3;
    float w = __expf(-10.0f * dist);
    ws += w;
    sd += w * dist;
  }
  float m = mask[row];
  wsum[row]  = ws * m;
  sdist[row] = sd * m;
}

// ---------------------------------------------------------------------------
// Kernel 2: build augmented features h (f16, stride LDH):
// h = [ x , x*(wsum-1) , sdist , 0 ... ]   (ss - xx_aug fused: x*wm - x)
// ---------------------------------------------------------------------------
__global__ void k_build_h(const float* __restrict__ xbuf, const float* __restrict__ wsum,
                          const float* __restrict__ sdist, int fin,
                          _Float16* __restrict__ h) {
  int idx = blockIdx.x * blockDim.x + threadIdx.x;
  if (idx >= Mrows * LDH) return;
  int row = idx / LDH, k = idx - row * LDH;
  float v;
  if (k < fin)               v = xbuf[(size_t)row * LDX + k];
  else if (k < 2 * fin)      v = xbuf[(size_t)row * LDX + (k - fin)] * (wsum[row] - 1.0f);
  else if (k == 2 * fin)     v = sdist[row];
  else                       v = 0.0f;
  h[idx] = (_Float16)v;
}

// ---------------------------------------------------------------------------
// Kernel 3: weight convert + transpose + zero-pad: f32 (Kreal x Nreal) row-major
// -> f16 Wt (Np x Kpad), Wt[n][k]. Zero pads guarantee padded-K products vanish.
// ---------------------------------------------------------------------------
__global__ void k_convert_wt(const float* __restrict__ W, int Kreal, int Nreal,
                             _Float16* __restrict__ wt, int Kpad, int Np) {
  int idx = blockIdx.x * blockDim.x + threadIdx.x;
  if (idx >= Np * Kpad) return;
  int n = idx / Kpad, k = idx - n * Kpad;
  float v = (k < Kreal && n < Nreal) ? W[(size_t)k * Nreal + n] : 0.0f;
  wt[idx] = (_Float16)v;
}

// ---------------------------------------------------------------------------
// Kernel 4: fused WMMA GEMM + epilogue.
// One wave computes one 16x16 tile of D = H(16xK) * Wt^T(Kx16) via
// v_wmma_f32_16x16x32_f16. 4 waves / block over consecutive M tiles.
// GNN mode:   y=(acc+bias)*mask; out = tanh(BN(y))  -> f32, stride ldout
// DENSE mode: y= acc+bias;       out = sigmoid(BN)  -> f16, stride ldout
// ---------------------------------------------------------------------------
template<bool DENSE>
__global__ __launch_bounds__(128)
void k_wmma_gemm(const _Float16* __restrict__ H, int ldh,
                 const _Float16* __restrict__ Wt, int ldw, int ksteps,
                 const float* __restrict__ bias,
                 const float* __restrict__ gamma, const float* __restrict__ beta,
                 const float* __restrict__ mean, const float* __restrict__ var,
                 const float* __restrict__ mask,
                 float* __restrict__ outF, _Float16* __restrict__ outH,
                 int ldout, int ncols) {
  const int lane = threadIdx.x & 31;
  const int wave = threadIdx.x >> 5;
  const int m0 = (blockIdx.x * 4 + wave) * 16;
  const int n0 = blockIdx.y * 16;

  // A fragment addressing (16-bit A 16x32): lanes 0-15 rows, K {0-7,16-23};
  // lanes 16-31 same rows, K {8-15,24-31}.
  const int arow = m0 + (lane & 15);
  const int aklo = (lane < 16) ? 0 : 8;
  // B fragment (32x16): lanes 0-15 col n, K {0-7,16-23}; lanes 16-31 col n, K {8-15,24-31}.
  const int bcol = n0 + (lane & 15);
  const int bkof = (lane < 16) ? 0 : 8;

  const _Float16* ap = H  + (size_t)arow * ldh + aklo;
  const _Float16* bp = Wt + (size_t)bcol * ldw + bkof;

  floatx8 acc = {};
  for (int ks = 0; ks < ksteps; ++ks) {
    half8 alo = *(const half8*)(ap);
    half8 ahi = *(const half8*)(ap + 16);
    half8 blo = *(const half8*)(bp);
    half8 bhi = *(const half8*)(bp + 16);
    half16 a, bfrag;
    #pragma unroll
    for (int i = 0; i < 8; ++i) {
      a[i] = alo[i];      a[i + 8] = ahi[i];
      bfrag[i] = blo[i];  bfrag[i + 8] = bhi[i];
    }
    acc = __builtin_amdgcn_wmma_f32_16x16x32_f16(
        /*neg_a=*/false, a, /*neg_b=*/false, bfrag,
        /*c_mod=*/(short)0, acc, /*reuse_a=*/false, /*reuse_b=*/false);
    ap += 32;
    bp += 32;
  }

  // D layout: lanes 0-15 col n, rows r..r+? : VGPR r -> M=r (lanes<16) / M=8+r.
  const int col = n0 + (lane & 15);
  const int rbase = m0 + ((lane >> 4) << 3);
  if (col < ncols) {
    const float bi = bias[col];
    const float g = gamma[col], be = beta[col], mu = mean[col];
    const float rs = rsqrtf(var[col] + EPSBN);
    #pragma unroll
    for (int r = 0; r < 8; ++r) {
      const int row = rbase + r;
      float y = acc[r] + bi;
      if (!DENSE) y *= mask[row];
      float t = g * (y - mu) * rs + be;
      if (DENSE) {
        float s = 1.0f / (1.0f + __expf(-t));
        outH[(size_t)row * ldout + col] = (_Float16)s;
      } else {
        outF[(size_t)row * ldout + col] = tanhf(t);
      }
    }
  }
}

// ---------------------------------------------------------------------------
// Kernel 5: f32 -> f16 (dense input, 256x4096 contiguous)
// ---------------------------------------------------------------------------
__global__ void k_f32tof16(const float* __restrict__ src, _Float16* __restrict__ dst, int n) {
  int idx = blockIdx.x * blockDim.x + threadIdx.x;
  if (idx < n) dst[idx] = (_Float16)src[idx];
}

// ---------------------------------------------------------------------------
// Kernel 6: final 200->2 layer + zero-pad to (256,4)
// ---------------------------------------------------------------------------
__global__ void k_final(const _Float16* __restrict__ act, const float* __restrict__ W2,
                        const float* __restrict__ b2, float* __restrict__ out) {
  int idx = blockIdx.x * blockDim.x + threadIdx.x;
  if (idx >= Bc * 4) return;
  int row = idx >> 2, c = idx & 3;
  if (c < 2) {
    float s = b2[c];
    const _Float16* a = act + (size_t)row * LDH;
    for (int j = 0; j < 200; ++j) s += (float)a[j] * W2[j * 2 + c];
    out[idx] = s;
  } else {
    out[idx] = 0.0f;
  }
}

// ---------------------------------------------------------------------------
// Host launcher
// ---------------------------------------------------------------------------
extern "C" void kernel_launch(void* const* d_in, const int* in_sizes, int n_in,
                              void* d_out, int out_size, void* d_ws, size_t ws_size,
                              hipStream_t stream) {
  (void)in_sizes; (void)n_in; (void)out_size; (void)ws_size;
  const float* xx   = (const float*)d_in[0];
  const int*   xcat = (const int*)d_in[1];
  const float* e0 = (const float*)d_in[2];
  const float* e1 = (const float*)d_in[3];
  const float* e2 = (const float*)d_in[4];
  const float* e3 = (const float*)d_in[5];

  // workspace carve-up (all sizes are 256B multiples)
  char* ws = (char*)d_ws;
  size_t off = 0;
  float*    xbuf  = (float*)(ws + off);    off += (size_t)Mrows * LDX * 4;   // 6,815,744
  float*    maskb = (float*)(ws + off);    off += (size_t)Mrows * 4;         // 65,536
  float*    wsum  = (float*)(ws + off);    off += (size_t)Mrows * 4;
  float*    sdist = (float*)(ws + off);    off += (size_t)Mrows * 4;
  _Float16* hgnn  = (_Float16*)(ws + off); off += (size_t)Mrows * LDH * 2;   // 7,340,032
  _Float16* wt    = (_Float16*)(ws + off); off += (size_t)208 * 4096 * 2;    // 1,703,936
  float*    flat  = (float*)(ws + off);    off += (size_t)Bc * 4096 * 4;     // 4,194,304
  _Float16* hdense= (_Float16*)(ws + off); off += (size_t)Bc * 4096 * 2;     // 2,097,152
  _Float16* dactA = (_Float16*)(ws + off); off += (size_t)Bc * LDH * 2;      // 114,688
  _Float16* dactB = (_Float16*)(ws + off); off += (size_t)Bc * LDH * 2;

  k_embed<<<(Mrows + 127) / 128, 128, 0, stream>>>(xx, xcat, e0, e1, e2, e3, xbuf, maskb);

  const int fins[5]  = {44, 100, 100, 100, 100};
  const int outsL[5] = {100, 100, 100, 100, 64};
  const int ndims[5] = {2, 4, 4, 4, 4};

  for (int L = 0; L < 5; ++L) {
    const int base = 6 + 6 * L;
    const float* A  = (const float*)d_in[base + 0];
    const float* bv = (const float*)d_in[base + 1];
    const float* ga = (const float*)d_in[base + 2];
    const float* be = (const float*)d_in[base + 3];
    const float* mu = (const float*)d_in[base + 4];
    const float* va = (const float*)d_in[base + 5];
    const int in_aug = 2 * fins[L] + 1;
    const int ksteps = (in_aug + 31) / 32;            // 3 or 7
    const int Np = ((outsL[L] + 15) / 16) * 16;       // 112 or 64

    k_pairwise<<<Bc, Nc, 0, stream>>>(xbuf, maskb, fins[L], ndims[L], wsum, sdist);
    k_build_h<<<(Mrows * LDH + 255) / 256, 256, 0, stream>>>(xbuf, wsum, sdist, fins[L], hgnn);
    k_convert_wt<<<(Np * LDH + 255) / 256, 256, 0, stream>>>(A, in_aug, outsL[L], wt, LDH, Np);

    float* oF = (L < 4) ? xbuf : flat;
    int ldo   = (L < 4) ? LDX : 64;                   // stride 64 == (B,4096) reshape
    dim3 grid(Mrows / 16 / 4, Np / 16);
    k_wmma_gemm<false><<<grid, 128, 0, stream>>>(hgnn, LDH, wt, LDH, ksteps,
                                                 bv, ga, be, mu, va, maskb,
                                                 oF, nullptr, ldo, outsL[L]);
  }

  k_f32tof16<<<(Bc * 4096 + 255) / 256, 256, 0, stream>>>(flat, hdense, Bc * 4096);

  const _Float16* hin = hdense;
  int ldh = 4096;
  _Float16* dst = dactA;
  for (int L = 0; L < 3; ++L) {
    const int base = 36 + 6 * L;
    const float* W  = (const float*)d_in[base + 0];
    const float* bv = (const float*)d_in[base + 1];
    const float* ga = (const float*)d_in[base + 2];
    const float* be = (const float*)d_in[base + 3];
    const float* mu = (const float*)d_in[base + 4];
    const float* va = (const float*)d_in[base + 5];
    const int Kreal = (L == 0) ? 4096 : 200;
    const int Kpad  = (L == 0) ? 4096 : LDH;
    const int ksteps = Kpad / 32;                     // 128 or 7

    k_convert_wt<<<(208 * Kpad + 255) / 256, 256, 0, stream>>>(W, Kreal, 200, wt, Kpad, 208);
    dim3 grid(Bc / 16 / 4, 208 / 16);                 // 4 x 13
    k_wmma_gemm<true><<<grid, 128, 0, stream>>>(hin, ldh, wt, Kpad, ksteps,
                                                bv, ga, be, mu, va, nullptr,
                                                nullptr, dst, LDH, 200);
    hin = dst;
    ldh = LDH;
    dst = (dst == dactA) ? dactB : dactA;
  }
  // after loop: last output is in dactA (A -> B -> A)
  k_final<<<(Bc * 4 + 255) / 256, 256, 0, stream>>>(dactA, (const float*)d_in[54],
                                                   (const float*)d_in[55], (float*)d_out);
}
